// PreprocessPositions_37572373906142
// MI455X (gfx1250) — matrix operand order repaired and checked by reading, not compiled, not run on hardware
//
#include <hip/hip_runtime.h>
#include <hip/hip_bf16.h>
#include <stdint.h>

// ---------------------------------------------------------------------------
// Problem sizes (fixed by the reference)
//   B=32, N=128, K=128, H=32, E=768
//   out = attn_bias [B,H,N,N] f32  ++  node_feature [B,N,E] f32
// ---------------------------------------------------------------------------

typedef __attribute__((ext_vector_type(16))) _Float16 v16h;
typedef __attribute__((ext_vector_type(8)))  _Float16 v8h;
typedef __attribute__((ext_vector_type(8)))  float    v8f;

// gcc-style int4 to match the async-to-LDS builtin prototype exactly
typedef int v4i_b __attribute__((vector_size(16)));
typedef __attribute__((address_space(1))) v4i_b gbuf_t;   // global (AS1)
typedef __attribute__((address_space(3))) v4i_b lbuf_t;   // LDS    (AS3)

#define DF_STRIDE 136   // halves per row of the 128x128 LDS tile (pads banks)

static __device__ inline v16h make_frag(v8h lo, v8h hi) {
  v16h r;
#pragma unroll
  for (int i = 0; i < 8; ++i) { r[i] = lo[i]; r[i + 8] = hi[i]; }
  return r;
}

// A-matrix fragment (16xK tile, row-major source with stride_h halves).
// ISA 16-bit A 16x32 layout: lane m=L&15; lanes<16 hold k = koff+{0..7, 16..23},
// lanes>=16 hold k = koff+8+{0..7, 16..23}.
static __device__ inline v16h load_a(const _Float16* base, int stride_h,
                                     int koff, int lane) {
  int m  = lane & 15;
  int kb = koff + ((lane & 16) ? 8 : 0);
  const _Float16* p = base + m * stride_h + kb;
  v8h lo = *(const v8h*)(p);
  v8h hi = *(const v8h*)(p + 16);
  return make_frag(lo, hi);
}

// B-matrix fragment from a PRE-TRANSPOSED weight wt[n][k] (stride_h halves).
// ISA 16-bit B 32x16 layout: lane n=L&15; lanes<16 hold k=koff+{0..7, 16..23},
// lanes>=16 hold k=koff+8+{0..7, 16..23}.
static __device__ inline v16h load_b(const _Float16* wt, int stride_h,
                                     int ncol0, int koff, int lane) {
  const _Float16* p = wt + (ncol0 + (lane & 15)) * stride_h + koff
                         + ((lane & 16) ? 8 : 0);
  v8h lo = *(const v8h*)(p);
  v8h hi = *(const v8h*)(p + 16);
  return make_frag(lo, hi);
}

static __device__ inline v8f wmma16(v16h a, v16h b, v8f c) {
  return __builtin_amdgcn_wmma_f32_16x16x32_f16(false, a, false, b,
                                                (short)0, c, false, false);
}

// Branch-free tanh: native V_TANH_F32 on gfx1250 if the builtin is declared,
// otherwise an exp-based identity that lowers to v_exp_f32 (no libm branches).
static __device__ inline float fast_tanh(float x) {
#if __has_builtin(__builtin_amdgcn_tanhf)
  return __builtin_amdgcn_tanhf(x);
#elif __has_builtin(__builtin_amdgcn_tanh_f32)
  return __builtin_amdgcn_tanh_f32(x);
#else
  return 1.0f - 2.0f / (1.0f + __expf(2.0f * x));
#endif
}

static __device__ inline float gelu_tanh(float x) {
  float t = fast_tanh(0.7978845608028654f * (x + 0.044715f * x * x * x));
  return 0.5f * x * (1.0f + t);
}

// ---------------------------------------------------------------------------
// Kernel 1: weight transpose + f16 convert, RBF constant folding
// ---------------------------------------------------------------------------
__global__ __launch_bounds__(256)
void prep_kernel(const float* __restrict__ means, const float* __restrict__ stds,
                 const float* __restrict__ w1, const float* __restrict__ w2,
                 const float* __restrict__ nodew,
                 _Float16* __restrict__ w1t, _Float16* __restrict__ w2t,
                 _Float16* __restrict__ nwt,
                 float* __restrict__ mu, float* __restrict__ inv_std,
                 float* __restrict__ coef) {
  int t = blockIdx.x * blockDim.x + threadIdx.x;
  if (t < 16384) {                       // w1t[l][k] = w1[k][l]
    int l = t >> 7, k = t & 127;
    w1t[l * 128 + k] = (_Float16)w1[k * 128 + l];
  } else if (t < 16384 + 4096) {         // w2t[h][k] = w2[k][h]
    int u = t - 16384;
    int h = u >> 7, k = u & 127;
    w2t[h * 128 + k] = (_Float16)w2[k * 32 + h];
  } else if (t < 16384 + 4096 + 98304) { // nwt[e][k] = node_w[k][e]
    int u = t - 20480;
    int e = u >> 7, k = u & 127;
    nwt[e * 128 + k] = (_Float16)nodew[k * 768 + e];
  } else if (t < 16384 + 4096 + 98304 + 128) {
    int k = t - 118784;
    float s = fabsf(stds[k]) + 0.01f;
    mu[k] = means[k];
    inv_std[k] = 1.0f / s;
    coef[k] = 0.3989422804014327f / s;   // 1/(sqrt(2*pi)*s)
  }
}

// ---------------------------------------------------------------------------
// Kernel 2: per-(b,i) fused RBF -> MLP -> attn_bias tile, plus df_sum row.
// Grid: 4096 blocks (b*128+i), 256 threads = 8 waves.
// ---------------------------------------------------------------------------
__global__ __launch_bounds__(256)
void attn_bias_kernel(const float* __restrict__ pos,
                      const unsigned char* __restrict__ mask,
                      const float* __restrict__ mu,
                      const float* __restrict__ inv_std,
                      const float* __restrict__ coef,
                      const _Float16* __restrict__ w1t,
                      const float* __restrict__ b1,
                      const _Float16* __restrict__ w2t,
                      const float* __restrict__ b2,
                      float* __restrict__ attn_out,
                      _Float16* __restrict__ dfsum_out) {
  __shared__ float s_d[128];
  __shared__ float s_maskf[128];
  __shared__ __align__(16) _Float16 s_buf[128 * DF_STRIDE];  // DF -> H1 -> attnT

  const int tid  = threadIdx.x;
  const int lane = tid & 31;
  const int wave = tid >> 5;
  const int blk  = blockIdx.x;
  const int b    = blk >> 7;
  const int i    = blk & 127;

  const float p00 = pos[(size_t)b * 128 * 3];
  const bool nanflag = (p00 != p00);

  // --- distances + mask ---------------------------------------------------
  if (tid < 128) {
    int j = tid;
    float xi = pos[((size_t)b * 128 + i) * 3 + 0];
    float yi = pos[((size_t)b * 128 + i) * 3 + 1];
    float zi = pos[((size_t)b * 128 + i) * 3 + 2];
    float xj = pos[((size_t)b * 128 + j) * 3 + 0];
    float yj = pos[((size_t)b * 128 + j) * 3 + 1];
    float zj = pos[((size_t)b * 128 + j) * 3 + 2];
    if (nanflag) { xi = yi = zi = 0.f; xj = yj = zj = 0.f; }
    float dx = xi - xj, dy = yi - yj, dz = zi - zj;
    float sq = dx * dx + dy * dy + dz * dz;
    s_d[j] = (sq == 0.0f) ? 0.0f : sqrtf(sq);
    s_maskf[j] = mask[(size_t)b * 128 + j] ? 1.0f : 0.0f;
  }
  __syncthreads();

  // --- RBF expansion: DF[j][k] as f16 in LDS ------------------------------
  {
    int k  = tid & 127;
    int j0 = tid >> 7;
    float muk = mu[k], isk = inv_std[k], ck = coef[k];
#pragma unroll 8
    for (int it = 0; it < 64; ++it) {
      int j = j0 + it * 2;
      float z = (s_d[j] - muk) * isk;
      s_buf[j * DF_STRIDE + k] = (_Float16)(__expf(-0.5f * z * z) * ck);
    }
  }
  __syncthreads();

  // --- masked df_sum row (f32 accumulate, stored f16 for node GEMM) ------
  if (tid < 128) {
    float s = 0.0f;
    for (int j = 0; j < 128; ++j)
      s += s_maskf[j] * (float)s_buf[j * DF_STRIDE + tid];
    dfsum_out[((size_t)b * 128 + i) * 128 + tid] = (_Float16)s;
  }
  __syncthreads();

  // --- GEMM1: H1 = gelu(DF @ w1 + b1), each wave owns 16 j-rows ----------
  const _Float16* arow = s_buf + wave * 16 * DF_STRIDE;
  const int nlo = lane & 15;
  const int moff = (lane & 16) ? 8 : 0;

  v16h aF[4];
#pragma unroll
  for (int ks = 0; ks < 4; ++ks) aF[ks] = load_a(arow, DF_STRIDE, ks * 32, lane);

#pragma unroll
  for (int lt = 0; lt < 8; ++lt) {
    v8f acc = {};
#pragma unroll
    for (int ks = 0; ks < 4; ++ks)
      acc = wmma16(aF[ks], load_b(w1t, 128, lt * 16, ks * 32, lane), acc);
    float bias = b1[lt * 16 + nlo];
#pragma unroll
    for (int v = 0; v < 8; ++v) {
      float g = gelu_tanh(acc[v] + bias);
      int j = wave * 16 + v + moff;   // rows owned by this wave only
      s_buf[j * DF_STRIDE + lt * 16 + nlo] = (_Float16)g;
    }
  }

  // --- GEMM2: attn = H1 @ w2 + b2 ----------------------------------------
  v16h cF[4];
#pragma unroll
  for (int ks = 0; ks < 4; ++ks) cF[ks] = load_a(arow, DF_STRIDE, ks * 32, lane);
  __syncthreads();                     // everyone has hoisted H1 frags
  float* attnT = (float*)s_buf;        // reuse LDS as [128 j][32 h] f32

#pragma unroll
  for (int nt = 0; nt < 2; ++nt) {
    v8f acc = {};
#pragma unroll
    for (int ks = 0; ks < 4; ++ks)
      acc = wmma16(cF[ks], load_b(w2t, 128, nt * 16, ks * 32, lane), acc);
    int h = nt * 16 + nlo;
    float bias = b2[h];
#pragma unroll
    for (int v = 0; v < 8; ++v) {
      int j = wave * 16 + v + moff;
      attnT[j * 32 + h] = acc[v] + bias;
    }
  }
  __syncthreads();

  // --- masked, coalesced output: attn_bias[b][h][i][j] --------------------
  size_t outbase = (size_t)b * 32 * 128 * 128 + (size_t)i * 128;
#pragma unroll
  for (int it = 0; it < 16; ++it) {
    int idx = tid + it * 256;
    int h = idx >> 7;
    int j = idx & 127;
    float v = attnT[j * 32 + h];
    v = (s_maskf[j] == 0.0f) ? -1000.0f : v;
    if (nanflag) v = 0.0f;
    attn_out[outbase + (size_t)h * 128 * 128 + j] = v;
  }
}

// ---------------------------------------------------------------------------
// Kernel 3: node_feature = df_sum @ node_w + node_b   ([4096,128]x[128,768])
// Grid: 256 blocks (16-row tiles), 256 threads = 8 waves, 6 n-tiles each.
// A-tile staged into LDS with the gfx1250 async global->LDS path if available.
// ---------------------------------------------------------------------------
__global__ __launch_bounds__(256)
void node_kernel(const _Float16* __restrict__ dfsum,
                 const _Float16* __restrict__ nwt,
                 const float* __restrict__ node_b,
                 const float* __restrict__ pos,
                 float* __restrict__ out) {
  const int tid  = threadIdx.x;
  const int lane = tid & 31;
  const int wave = tid >> 5;
  const int row0 = blockIdx.x * 16;
  const int b    = row0 >> 7;
  const float p00 = pos[(size_t)b * 128 * 3];
  const bool nanflag = (p00 != p00);
  const int nlo = lane & 15;
  const int moff = (lane & 16) ? 8 : 0;

  v16h aF[4];

#if __has_builtin(__builtin_amdgcn_global_load_async_to_lds_b128)
  __shared__ __align__(16) _Float16 s_a[16 * 128];           // 4 KB A-tile
  {
    // 256 lanes x 16B = 4096B = whole 16x128 f16 tile, one async op per lane.
    // Casts go through uintptr_t: AS1 keeps the 64-bit pattern, AS3 is the
    // low-32-bit LDS offset (flat->LDS mapping uses addr[31:0]).
    const _Float16* g = dfsum + (size_t)row0 * 128 + tid * 8;
    gbuf_t* gp = (gbuf_t*)(uintptr_t)g;
    lbuf_t* lp = (lbuf_t*)(uint32_t)(uintptr_t)(s_a + tid * 8);
    __builtin_amdgcn_global_load_async_to_lds_b128(gp, lp, 0, 0);
  }
#if __has_builtin(__builtin_amdgcn_s_wait_asynccnt)
  __builtin_amdgcn_s_wait_asynccnt(0);
#else
  asm volatile("s_wait_asynccnt 0x0" ::: "memory");
#endif
  __syncthreads();
#pragma unroll
  for (int ks = 0; ks < 4; ++ks)
    aF[ks] = load_a(s_a, 128, ks * 32, lane);
#else
#pragma unroll
  for (int ks = 0; ks < 4; ++ks)
    aF[ks] = load_a(dfsum + (size_t)row0 * 128, 128, ks * 32, lane);
#endif

#pragma unroll
  for (int t = 0; t < 6; ++t) {
    int nt = wave + t * 8;             // 0..47 n-tiles of E=768
    v8f acc = {};
#pragma unroll
    for (int ks = 0; ks < 4; ++ks)
      acc = wmma16(aF[ks], load_b(nwt, 128, nt * 16, ks * 32, lane), acc);
    int e = nt * 16 + nlo;
    float bias = node_b[e];
#pragma unroll
    for (int v = 0; v < 8; ++v) {
      int r = row0 + v + moff;
      float val = acc[v] + bias;
      out[(size_t)r * 768 + e] = nanflag ? 0.0f : val;
    }
  }
}

// ---------------------------------------------------------------------------
// Host launcher
// ---------------------------------------------------------------------------
extern "C" void kernel_launch(void* const* d_in, const int* in_sizes, int n_in,
                              void* d_out, int out_size, void* d_ws, size_t ws_size,
                              hipStream_t stream) {
  (void)in_sizes; (void)n_in; (void)out_size; (void)ws_size;

  const float*         pos    = (const float*)d_in[0];
  const unsigned char* mask   = (const unsigned char*)d_in[1];
  const float*         means  = (const float*)d_in[2];
  const float*         stds   = (const float*)d_in[3];
  const float*         w1     = (const float*)d_in[4];
  const float*         b1     = (const float*)d_in[5];
  const float*         w2     = (const float*)d_in[6];
  const float*         b2     = (const float*)d_in[7];
  const float*         node_w = (const float*)d_in[8];
  const float*         node_b = (const float*)d_in[9];

  char* ws = (char*)d_ws;
  _Float16* w1t     = (_Float16*)(ws);             // 32768 B
  _Float16* w2t     = (_Float16*)(ws + 32768);     //  8192 B
  _Float16* nwt     = (_Float16*)(ws + 40960);     // 196608 B
  float*    mu      = (float*)(ws + 237568);       //   512 B
  float*    inv_std = (float*)(ws + 238080);       //   512 B
  float*    coef    = (float*)(ws + 238592);       //   512 B
  _Float16* dfsum   = (_Float16*)(ws + 239616);    // 1 MiB (4096*128 f16)

  float* attn_out = (float*)d_out;                             // [32,32,128,128]
  float* node_out = (float*)d_out + (size_t)32 * 32 * 128 * 128; // [32,128,768]

  prep_kernel<<<465, 256, 0, stream>>>(means, stds, w1, w2, node_w,
                                       w1t, w2t, nwt, mu, inv_std, coef);

  attn_bias_kernel<<<4096, 256, 0, stream>>>(pos, mask, mu, inv_std, coef,
                                             w1t, b1, w2t, b2,
                                             attn_out, dfsum);

  node_kernel<<<256, 256, 0, stream>>>(dfsum, nwt, node_b, pos, node_out);
}
